// TransformerLayerDecoderMH_76682346103552
// MI455X (gfx1250) — compile-verified, matrix operand order checked by reading, not run on hardware
//
#include <hip/hip_runtime.h>

// ---------------------------------------------------------------------------
// Problem constants (from reference)
// ---------------------------------------------------------------------------
#define BB    2
#define TSRC  2048
#define TTGT  2048
#define DMOD  1024
#define DINN  1024
#define NHEAD 16
#define DHEAD 64
#define DFF   2048
#define NTOK  (BB * TTGT)   // 4096 flattened decoder tokens
#define NSRC  (BB * TSRC)   // 4096 flattened encoder tokens

typedef __attribute__((ext_vector_type(16))) __bf16 v16bf;
typedef __attribute__((ext_vector_type(8)))  float  v8f;

// ---------------------------------------------------------------------------
// Fragment loader: two aligned 16-byte chunks -> one 16-element bf16 fragment
// CDNA5 16-bit A/B WMMA striping: halves 0..7 = K kb..kb+7, halves 8..15 =
// K kb+16..kb+23 (kb = 0 for lanes 0-15, 8 for lanes 16-31).
// ---------------------------------------------------------------------------
__device__ static inline v16bf ldfrag(const __bf16* p0, const __bf16* p1) {
  union U { uint4 u; __bf16 h[8]; };
  U a, b;
  a.u = *(const uint4*)p0;
  b.u = *(const uint4*)p1;
  v16bf f;
#pragma unroll
  for (int i = 0; i < 8; ++i) { f[i] = a.h[i]; f[8 + i] = b.h[i]; }
  return f;
}

__device__ static inline float rmax16(float v) {
#pragma unroll
  for (int m = 1; m < 16; m <<= 1) v = fmaxf(v, __shfl_xor(v, m, 32));
  return v;
}
__device__ static inline float rsum16(float v) {
#pragma unroll
  for (int m = 1; m < 16; m <<= 1) v += __shfl_xor(v, m, 32);
  return v;
}

// LDS byte offset (workgroup-relative) of a generic pointer into __shared__:
// flat LDS-aperture addresses carry the LDS offset in the low 32 bits.
__device__ static inline unsigned lds_off(const void* p) {
  return (unsigned)(unsigned long long)p;
}

// ---------------------------------------------------------------------------
// bf16 GEMM:  out = op(A[MxK] * B[KxN] + bias) [+ res],  op = optional relu
// 128x128 block tile, 4 waves (wave32), each wave 64x64 via 4x4 WMMA frags.
// Double-buffered LDS; A tile staged with GLOBAL_LOAD_ASYNC_TO_LDS_B128
// (ASYNCcnt), B tile staged manually because it is transposed to [n][k].
// ---------------------------------------------------------------------------
template <bool RELU, bool RES, bool OUTF, bool OUTB>
__global__ __launch_bounds__(128) void gemm_bf16_wmma(
    const __bf16* __restrict__ A, const __bf16* __restrict__ Bm,
    const float* __restrict__ bias, const float* __restrict__ res,
    float* __restrict__ outf, __bf16* __restrict__ outb,
    int M, int N, int K)
{
  __shared__ __align__(16) __bf16 As[2][128][40];  // [buf][m][k], 80B stride
  __shared__ __align__(16) __bf16 Bs[2][128][40];  // [buf][n][k]

  const int tid  = threadIdx.x;
  const int lane = tid & 31;
  const int wid  = tid >> 5;
  const int wm   = wid >> 1, wn = wid & 1;
  const int m0   = blockIdx.y * 128, n0 = blockIdx.x * 128;
  const int lr   = lane & 15;
  const int kb   = (lane >= 16) ? 8 : 0;
  const int radd = (lane >= 16) ? 8 : 0;

  auto stage = [&](int buf, int kt) {
    const int k0 = kt << 5;
    // A tile: async global->LDS, 16B per lane per issue (no VGPR round trip)
#pragma unroll
    for (int it = 0; it < 4; ++it) {
      int idx = tid + it * 128;            // 0..511
      int row = idx >> 2, kg = idx & 3;
      unsigned            l = lds_off(&As[buf][row][kg * 8]);
      unsigned long long  g = (unsigned long long)
          &A[(size_t)(m0 + row) * K + k0 + kg * 8];
      asm volatile("global_load_async_to_lds_b128 %0, %1, off"
                   :: "v"(l), "v"(g) : "memory");
    }
    // B tile transposed to [n][k] (manual: async copies cannot transpose)
#pragma unroll
    for (int it = 0; it < 4; ++it) {
      int idx = tid + it * 128;
      int k = idx >> 4, ng = idx & 15;
      union { uint4 u; __bf16 h[8]; } t;
      t.u = *(const uint4*)&Bm[(size_t)(k0 + k) * N + n0 + ng * 8];
#pragma unroll
      for (int j = 0; j < 8; ++j) Bs[buf][ng * 8 + j][k] = t.h[j];
    }
  };

  v8f acc[4][4];
#pragma unroll
  for (int fm = 0; fm < 4; ++fm)
#pragma unroll
    for (int fn = 0; fn < 4; ++fn)
#pragma unroll
      for (int i = 0; i < 8; ++i) acc[fm][fn][i] = 0.f;

  const int KT = K >> 5;
  stage(0, 0);
  asm volatile("s_wait_asynccnt 0x0" ::: "memory");
  __syncthreads();

  for (int kt = 0; kt < KT; ++kt) {
    const int cur = kt & 1;
    if (kt + 1 < KT) stage(1 - cur, kt + 1);   // overlap with compute

    v16bf af[4], bf[4];
#pragma unroll
    for (int fm = 0; fm < 4; ++fm) {
      const __bf16* p = &As[cur][wm * 64 + fm * 16 + lr][kb];
      af[fm] = ldfrag(p, p + 16);
    }
#pragma unroll
    for (int fn = 0; fn < 4; ++fn) {
      const __bf16* p = &Bs[cur][wn * 64 + fn * 16 + lr][kb];
      bf[fn] = ldfrag(p, p + 16);
    }
#pragma unroll
    for (int fm = 0; fm < 4; ++fm)
#pragma unroll
      for (int fn = 0; fn < 4; ++fn)
        acc[fm][fn] = __builtin_amdgcn_wmma_f32_16x16x32_bf16(
            false, af[fm], false, bf[fn], (short)0, acc[fm][fn], false, false);

    // retire this wave's async loads, then workgroup barrier
    asm volatile("s_wait_asynccnt 0x0" ::: "memory");
    __syncthreads();
  }

  // epilogue (branch-free: template-specialized)
#pragma unroll
  for (int fn = 0; fn < 4; ++fn) {
    const int col = n0 + wn * 64 + fn * 16 + lr;
    const float bcol = bias[col];
#pragma unroll
    for (int fm = 0; fm < 4; ++fm)
#pragma unroll
      for (int i = 0; i < 8; ++i) {
        int row = m0 + wm * 64 + fm * 16 + i + radd;
        float c = acc[fm][fn][i] + bcol;
        if (RELU) c = fmaxf(c, 0.f);
        size_t o = (size_t)row * N + col;
        if (RES)  c += res[o];
        if (OUTF) outf[o] = c;
        if (OUTB) outb[o] = (__bf16)c;
      }
  }
}

// ---------------------------------------------------------------------------
// Flash-style attention: one block = 64 query rows x 1 head x 1 batch,
// 4 waves, each wave owns 16 query rows. K/V chunks of 64 staged in LDS.
// S = Q*K^T and O += P*V both via v_wmma_f32_16x16x32_bf16.
// ---------------------------------------------------------------------------
__global__ __launch_bounds__(128) void attn_wmma(
    const __bf16* __restrict__ Q, const __bf16* __restrict__ Kg,
    const __bf16* __restrict__ Vg, __bf16* __restrict__ O,
    int Tq, int Tk, int causal)
{
  __shared__ __align__(16) __bf16 Ks[64][72];      // [key][dh]
  __shared__ __align__(16) __bf16 Vs[64][72];      // [dh][key] (transposed)
  __shared__ __align__(16) __bf16 Ps[4][16][72];   // per-wave P scratch

  const int tid  = threadIdx.x, lane = tid & 31, w = tid >> 5;
  const int lr   = lane & 15;
  const int kb   = (lane >= 16) ? 8 : 0;
  const int radd = (lane >= 16) ? 8 : 0;
  const int q0   = blockIdx.x * 64;
  const int h    = blockIdx.y;
  const int b    = blockIdx.z;

  const size_t qrow = ((size_t)b * Tq + q0 + w * 16 + lr) * DINN + h * DHEAD;
  v16bf qf[2];
#pragma unroll
  for (int ks = 0; ks < 2; ++ks)
    qf[ks] = ldfrag(&Q[qrow + ks * 32 + kb], &Q[qrow + ks * 32 + kb + 16]);

  float mrow[8], lrow[8];
  v8f of[4];
#pragma unroll
  for (int i = 0; i < 8; ++i) { mrow[i] = -3.0e38f; lrow[i] = 0.f; }
#pragma unroll
  for (int t = 0; t < 4; ++t)
#pragma unroll
    for (int i = 0; i < 8; ++i) of[t][i] = 0.f;

  const int cend = causal ? ((int)blockIdx.x + 1) : (Tk >> 6);
  for (int c = 0; c < cend; ++c) {
    const int kbase = c * 64;
    __syncthreads();
#pragma unroll
    for (int it = 0; it < 4; ++it) {
      int idx = tid + it * 128;            // 0..511
      int row = idx >> 3, dg = idx & 7;
      *(uint4*)&Ks[row][dg * 8] =
          *(const uint4*)&Kg[((size_t)b * Tk + kbase + row) * DINN + h * DHEAD + dg * 8];
    }
#pragma unroll
    for (int it = 0; it < 4; ++it) {
      int idx = tid + it * 128;
      int key = idx >> 3, dg = idx & 7;
      union { uint4 u; __bf16 hh[8]; } t;
      t.u = *(const uint4*)&Vg[((size_t)b * Tk + kbase + key) * DINN + h * DHEAD + dg * 8];
#pragma unroll
      for (int j = 0; j < 8; ++j) Vs[dg * 8 + j][key] = t.hh[j];
    }
    __syncthreads();

    // S = (Q K^T) / sqrt(DH), masked
    float p[4][8];
#pragma unroll
    for (int t = 0; t < 4; ++t) {
      v8f s;
#pragma unroll
      for (int i = 0; i < 8; ++i) s[i] = 0.f;
#pragma unroll
      for (int ks = 0; ks < 2; ++ks) {
        const __bf16* kp = &Ks[t * 16 + lr][ks * 32 + kb];
        v16bf bfK = ldfrag(kp, kp + 16);
        s = __builtin_amdgcn_wmma_f32_16x16x32_bf16(
            false, qf[ks], false, bfK, (short)0, s, false, false);
      }
#pragma unroll
      for (int i = 0; i < 8; ++i) {
        float val = s[i] * 0.125f;                 // 1/sqrt(64)
        int col  = kbase + t * 16 + lr;
        int qid  = q0 + w * 16 + i + radd;
        if (causal && col > qid) val = -3.0e38f;
        p[t][i] = val;
      }
    }

    // online softmax update (per-lane rows: i + radd)
#pragma unroll
    for (int i = 0; i < 8; ++i) {
      float nm = mrow[i];
#pragma unroll
      for (int t = 0; t < 4; ++t) nm = fmaxf(nm, rmax16(p[t][i]));
      float alpha = __expf(mrow[i] - nm);
      float acc = 0.f;
#pragma unroll
      for (int t = 0; t < 4; ++t) {
        float e = __expf(p[t][i] - nm);
        p[t][i] = e;
        acc += e;
      }
      acc = rsum16(acc);
      lrow[i] = lrow[i] * alpha + acc;
      mrow[i] = nm;
#pragma unroll
      for (int t = 0; t < 4; ++t) of[t][i] *= alpha;
#pragma unroll
      for (int t = 0; t < 4; ++t)
        Ps[w][i + radd][t * 16 + lr] = (__bf16)p[t][i];
    }

    // O += P @ V  (wave-private P; same-wave LDS ops are in order)
#pragma unroll
    for (int ks = 0; ks < 2; ++ks) {
      const __bf16* pp = &Ps[w][lr][ks * 32 + kb];
      v16bf pa = ldfrag(pp, pp + 16);
#pragma unroll
      for (int t = 0; t < 4; ++t) {
        const __bf16* vp = &Vs[t * 16 + lr][ks * 32 + kb];
        v16bf bv = ldfrag(vp, vp + 16);
        of[t] = __builtin_amdgcn_wmma_f32_16x16x32_bf16(
            false, pa, false, bv, (short)0, of[t], false, false);
      }
    }
  }

  // normalize and write bf16 output (B,T,DI)
#pragma unroll
  for (int t = 0; t < 4; ++t)
#pragma unroll
    for (int i = 0; i < 8; ++i) {
      int qid = q0 + w * 16 + i + radd;
      O[((size_t)b * Tq + qid) * DINN + h * DHEAD + t * 16 + lr] =
          (__bf16)(of[t][i] / lrow[i]);
    }
}

// ---------------------------------------------------------------------------
// Row LayerNorm over DM=1024; one block per row. Writes fp32 and/or bf16.
// ---------------------------------------------------------------------------
__global__ __launch_bounds__(256) void ln_kernel(
    const float* __restrict__ in, const float* __restrict__ g,
    const float* __restrict__ be, float* __restrict__ outf,
    __bf16* __restrict__ outb)
{
  const int row = blockIdx.x;
  const int tid = threadIdx.x;
  const float* p = in + (size_t)row * DMOD;
  float v[4], s = 0.f, s2 = 0.f;
#pragma unroll
  for (int i = 0; i < 4; ++i) {
    v[i] = p[tid + i * 256];
    s += v[i]; s2 += v[i] * v[i];
  }
  __shared__ float rs[256], rq[256];
  rs[tid] = s; rq[tid] = s2;
  __syncthreads();
  for (int off = 128; off > 0; off >>= 1) {
    if (tid < off) { rs[tid] += rs[tid + off]; rq[tid] += rq[tid + off]; }
    __syncthreads();
  }
  const float mean = rs[0] * (1.0f / DMOD);
  const float var  = rq[0] * (1.0f / DMOD) - mean * mean;
  const float inv  = rsqrtf(var + 1e-5f);
#pragma unroll
  for (int i = 0; i < 4; ++i) {
    int j = tid + i * 256;
    float y = (v[i] - mean) * inv * g[j] + be[j];
    if (outf) outf[(size_t)row * DMOD + j] = y;
    if (outb) outb[(size_t)row * DMOD + j] = (__bf16)y;
  }
}

// ---------------------------------------------------------------------------
// fp32 -> bf16 conversion helpers
// ---------------------------------------------------------------------------
__global__ void cvt_bf16_kernel(const float* __restrict__ in,
                                __bf16* __restrict__ out, int n) {
  int i = blockIdx.x * 256 + threadIdx.x;
  if (i < n) out[i] = (__bf16)in[i];
}
__global__ void copy_cvt_kernel(const float* __restrict__ in,
                                float* __restrict__ of,
                                __bf16* __restrict__ ob, int n) {
  int i = blockIdx.x * 256 + threadIdx.x;
  if (i < n) { float v = in[i]; of[i] = v; ob[i] = (__bf16)v; }
}

// ---------------------------------------------------------------------------
// Host-side orchestration
// ---------------------------------------------------------------------------
static inline void* ws_alloc(char* base, size_t& off, size_t bytes) {
  off = (off + 255) & ~(size_t)255;
  void* p = base + off;
  off += bytes;
  return p;
}

extern "C" void kernel_launch(void* const* d_in, const int* in_sizes, int n_in,
                              void* d_out, int out_size, void* d_ws, size_t ws_size,
                              hipStream_t stream) {
  (void)in_sizes; (void)n_in; (void)out_size; (void)ws_size;
  const float* enc  = (const float*)d_in[0];
  const float* outv = (const float*)d_in[1];
  // per-tag input base: dd=2, ed=18; order within tag:
  // Wq bq Wk bk Wv bv Wo bo W1 b1 W2 b2 g1 be1 g2 be2
  auto F = [&](int i) { return (const float*)d_in[i]; };

  char* base = (char*)d_ws;
  size_t off = 0;
  const size_t WQK = (size_t)DMOD * DINN;     // 1M elems
  const size_t WFF = (size_t)DMOD * DFF;      // 2M elems

  __bf16* wb[12];  // bf16 weights: dd{Wq,Wk,Wv,Wo,W1,W2}, ed{...}
  const int widx[12]  = {2, 4, 6, 8, 10, 12, 18, 20, 22, 24, 26, 28};
  const size_t wsz[12] = {WQK, WQK, WQK, WQK, WFF, WFF,
                          WQK, WQK, WQK, WQK, WFF, WFF};
  for (int i = 0; i < 12; ++i)
    wb[i] = (__bf16*)ws_alloc(base, off, wsz[i] * sizeof(__bf16));

  float*  x    = (float*)ws_alloc(base, off, (size_t)NTOK * DMOD * 4);
  float*  y    = (float*)ws_alloc(base, off, (size_t)NTOK * DMOD * 4);
  __bf16* xb   = (__bf16*)ws_alloc(base, off, (size_t)NTOK * DMOD * 2);
  __bf16* encb = (__bf16*)ws_alloc(base, off, (size_t)NSRC * DMOD * 2);
  __bf16* Qb   = (__bf16*)ws_alloc(base, off, (size_t)NTOK * DINN * 2);
  __bf16* Kb   = (__bf16*)ws_alloc(base, off, (size_t)NSRC * DINN * 2);
  __bf16* Vb   = (__bf16*)ws_alloc(base, off, (size_t)NSRC * DINN * 2);
  __bf16* Ab   = (__bf16*)ws_alloc(base, off, (size_t)NTOK * DINN * 2);
  __bf16* Hb   = (__bf16*)ws_alloc(base, off, (size_t)NTOK * DFF  * 2);

  // --- conversions ---
  copy_cvt_kernel<<<(NTOK * DMOD) / 256, 256, 0, stream>>>(outv, x, xb, NTOK * DMOD);
  cvt_bf16_kernel<<<(NSRC * DMOD) / 256, 256, 0, stream>>>(enc, encb, NSRC * DMOD);
  for (int i = 0; i < 12; ++i)
    cvt_bf16_kernel<<<(int)(wsz[i] / 256), 256, 0, stream>>>(F(widx[i]), wb[i], (int)wsz[i]);

  // GEMM variants (template-specialized epilogues)
  auto gemm_b16 = [&](const __bf16* A, const __bf16* B, const float* bias,
                      __bf16* ob, int M, int N, int K) {
    dim3 g(N / 128, M / 128);
    gemm_bf16_wmma<false, false, false, true><<<g, 128, 0, stream>>>(
        A, B, bias, nullptr, nullptr, ob, M, N, K);
  };
  auto gemm_relu_b16 = [&](const __bf16* A, const __bf16* B, const float* bias,
                           __bf16* ob, int M, int N, int K) {
    dim3 g(N / 128, M / 128);
    gemm_bf16_wmma<true, false, false, true><<<g, 128, 0, stream>>>(
        A, B, bias, nullptr, nullptr, ob, M, N, K);
  };
  auto gemm_res_f32 = [&](const __bf16* A, const __bf16* B, const float* bias,
                          const float* res, float* of, int M, int N, int K) {
    dim3 g(N / 128, M / 128);
    gemm_bf16_wmma<false, true, true, false><<<g, 128, 0, stream>>>(
        A, B, bias, res, of, nullptr, M, N, K);
  };
  auto ln = [&](const float* in, const float* g1, const float* b1,
                float* of, __bf16* ob) {
    ln_kernel<<<NTOK, 256, 0, stream>>>(in, g1, b1, of, ob);
  };
  dim3 ag(TTGT / 64, NHEAD, BB);

  // ================= self-attention block (dd) =================
  gemm_b16(xb, wb[0], F(3), Qb, NTOK, DINN, DMOD);                 // Q
  gemm_b16(xb, wb[1], F(5), Kb, NTOK, DINN, DMOD);                 // K
  gemm_b16(xb, wb[2], F(7), Vb, NTOK, DINN, DMOD);                 // V
  attn_wmma<<<ag, 128, 0, stream>>>(Qb, Kb, Vb, Ab, TTGT, TTGT, 1);
  gemm_res_f32(Ab, wb[3], F(9), x, y, NTOK, DMOD, DINN);           // Wo + res
  ln(y, F(14), F(15), x, xb);                                      // LN1
  gemm_relu_b16(xb, wb[4], F(11), Hb, NTOK, DFF, DMOD);            // FFN1+relu
  gemm_res_f32(Hb, wb[5], F(13), x, y, NTOK, DMOD, DFF);           // FFN2 + res
  ln(y, F(16), F(17), x, xb);                                      // LN2

  // ================= cross-attention block (ed) =================
  gemm_b16(xb,   wb[6], F(19), Qb, NTOK, DINN, DMOD);              // Q
  gemm_b16(encb, wb[7], F(21), Kb, NSRC, DINN, DMOD);              // K (enc)
  gemm_b16(encb, wb[8], F(23), Vb, NSRC, DINN, DMOD);              // V (enc)
  attn_wmma<<<ag, 128, 0, stream>>>(Qb, Kb, Vb, Ab, TTGT, TSRC, 0);
  gemm_res_f32(Ab, wb[9], F(25), x, y, NTOK, DMOD, DINN);          // Wo + res
  ln(y, F(30), F(31), x, xb);                                      // LN1
  gemm_relu_b16(xb, wb[10], F(27), Hb, NTOK, DFF, DMOD);           // FFN1+relu
  gemm_res_f32(Hb, wb[11], F(29), x, y, NTOK, DMOD, DFF);          // FFN2 + res
  ln_kernel<<<NTOK, 256, 0, stream>>>(y, F(32), F(33), (float*)d_out, nullptr);
}
